// Head_86663850099249
// MI455X (gfx1250) — matrix operand order repaired and checked by reading, not compiled, not run on hardware
//
#include <hip/hip_runtime.h>
#include <hip/hip_bf16.h>

// ---------------------------------------------------------------------------
// Causal attention head for MI455X (gfx1250, wave32, WMMA bf16 16x16x32).
//
// Phase 0 (wprep_kernel): one-time f32->bf16 conversion of Wq/Wk/Wv into
//                         WMMA B-layout tiles (they are reused by all 1024
//                         row tiles; converting once removes ~1000x redundant
//                         strided loads + cvts from the projection loop).
// Phase 1 (proj_kernel):  q,k,v = x @ W  (bf16 out, pre-swizzled into WMMA
//                         A/B register layouts (contiguous k = 16h + j) in d_ws).
//                         Softmax scale 1/sqrt(H) folded into q.
// Phase 2 (attn_kernel):  flash-attention with online softmax; one wave per
//                         16-row query tile; S and PV both via
//                         v_wmma_f32_16x16x32_bf16; P transposed D->A layout
//                         through per-wave LDS (ds stores + s_wait_dscnt);
//                         next k/v chunk prefetched via global_prefetch_b8.
//
// d_ws requirement: 12.75 MiB (q:4MiB, k:4MiB, v:4MiB, W-tiles:768KiB).
// ---------------------------------------------------------------------------

#define Bq 8
#define Tq 2048
#define Cq 1024
#define Hq 128

typedef __attribute__((ext_vector_type(16))) __bf16 v16bf;
typedef __attribute__((ext_vector_type(8)))  float  v8f;

// Tile sizes in bf16 elements
#define TILE_ELEMS 512                                   // one 16x32 / 32x16 tile
#define QHALVES   ((size_t)Bq * 128 * 4 * TILE_ELEMS)    // 2,097,152 halves (4 MiB)
#define KOFF      (QHALVES)
#define VOFF      (2 * QHALVES)
#define WOFF      (3 * QHALVES)                          // 3*32*8*512 halves of W tiles

// --- WMMA register layouts used here (contiguous k = 16h + s, 16-bit operands) ----
// A-matrix 16x32 (MxK): element (M,K) lives at lane = M + 16*(K>>4),
//                       half-index j = K & 15.
__device__ __forceinline__ int laneA(int M, int K) { return M + 16 * (K >> 4); }
__device__ __forceinline__ int jA(int K)           { return K & 15; }
// B-matrix 32x16 (KxN): element (K,N) lives at lane = N + 16*(K>>4),
//                       half-index j = K & 15.
__device__ __forceinline__ int laneB(int K, int N) { return N + 16 * (K >> 4); }
__device__ __forceinline__ int jB(int K)           { return K & 15; }

__device__ __forceinline__ v8f wmma_bf16(v16bf a, v16bf b, v8f c) {
  // (neg_a, A, neg_b, B, c_mod, C, reuse_a, reuse_b)
  return __builtin_amdgcn_wmma_f32_16x16x32_bf16(false, a, false, b, (short)0, c,
                                                 false, false);
}

// ---------------------------------------------------------------------------
// Phase 0: convert W matrices into bf16 B-layout tiles.
// Tile index (m, kc 0..31, n 0..7): K = C-chunk row (32), N = h%16 within
// h-tile n. 768 tiles, one wave each. grid = 192 blocks * 4 waves.
// ---------------------------------------------------------------------------
__global__ void __launch_bounds__(128)
wprep_kernel(const float* __restrict__ Wq, const float* __restrict__ Wk,
             const float* __restrict__ Wv, __bf16* __restrict__ qkv) {
  const int lane = threadIdx.x & 31;
  const int wave = threadIdx.x >> 5;
  const int id   = blockIdx.x * 4 + wave;     // 0..767
  const int m    = id >> 8;                   // 0..2
  const int kc   = (id & 255) >> 3;           // 0..31
  const int n    = id & 7;                    // 0..7
  const int g    = lane >> 4;
  const int Nl   = lane & 15;

  const float* W = (m == 0) ? Wq : ((m == 1) ? Wk : Wv);
  // Lane holds column h = 16*n + Nl, rows K = 16*g + j  (j = 0..15).
  const float* wp = W + (size_t)(kc * 32 + 16 * g) * Hq + n * 16 + Nl;
  v16bf bm;
  #pragma unroll
  for (int j = 0; j < 16; ++j) bm[j] = (__bf16)wp[(size_t)j * Hq];
  *(v16bf*)(qkv + WOFF + (((size_t)(m * 32 + kc) * 8 + n) << 9) + lane * 16) = bm;
}

// ---------------------------------------------------------------------------
// Phase 1: projections. grid = (1024 row-tiles, 3 matrices), block = 128
// (4 waves). Wave w owns output N-tiles {2w, 2w+1} (each 16 of H=128 cols).
// ---------------------------------------------------------------------------
__global__ void __launch_bounds__(128)
proj_kernel(const float* __restrict__ x, __bf16* __restrict__ qkv) {
  const int lane = threadIdx.x & 31;
  const int wave = threadIdx.x >> 5;
  const int tile = blockIdx.x;          // 0..1023 row tile over B*T
  const int m    = blockIdx.y;          // 0=q, 1=k, 2=v
  const int b    = tile >> 7;           // 128 tiles per batch
  const int ti   = tile & 127;
  const int t0   = ti << 4;
  const int n0   = wave << 1;           // first of two 16-col N-tiles
  const int g    = lane >> 4;           // half-wave group
  const int Nl   = lane & 15;

  v8f acc0 = {};
  v8f acc1 = {};

  // This lane's A-row of x (row M = Nl of the tile): K = 16*g + j
  // (j = 0..15)  -> four float4 per K-chunk.
  const float* xrow = x + ((size_t)(b * Tq + t0 + Nl)) * Cq;
  const __bf16* wtiles = qkv + WOFF + (((size_t)m * 32) << 12);  // m*32*8*512

  for (int kc = 0; kc < 32; ++kc) {
    const float* xp = xrow + kc * 32 + 16 * g;
    float4 x0 = *(const float4*)(xp + 0);
    float4 x1 = *(const float4*)(xp + 4);
    float4 x2 = *(const float4*)(xp + 8);
    float4 x3 = *(const float4*)(xp + 12);
    v16bf a;
    a[0]  = (__bf16)x0.x;  a[1]  = (__bf16)x0.y;
    a[2]  = (__bf16)x0.z;  a[3]  = (__bf16)x0.w;
    a[4]  = (__bf16)x1.x;  a[5]  = (__bf16)x1.y;
    a[6]  = (__bf16)x1.z;  a[7]  = (__bf16)x1.w;
    a[8]  = (__bf16)x2.x;  a[9]  = (__bf16)x2.y;
    a[10] = (__bf16)x2.z;  a[11] = (__bf16)x2.w;
    a[12] = (__bf16)x3.x;  a[13] = (__bf16)x3.y;
    a[14] = (__bf16)x3.z;  a[15] = (__bf16)x3.w;

    // Pre-swizzled bf16 B tiles: contiguous 32B per lane.
    v16bf b0 = *(const v16bf*)(wtiles + (((size_t)kc * 8 + n0 + 0) << 9) + lane * 16);
    v16bf b1 = *(const v16bf*)(wtiles + (((size_t)kc * 8 + n0 + 1) << 9) + lane * 16);
    acc0 = wmma_bf16(a, b0, acc0);
    acc1 = wmma_bf16(a, b1, acc1);
  }

  // Scatter-store into WMMA-ready bf16 tiles.
  const float qscale = 0.08838834764831845f;   // 1/sqrt(128), folded into q
  #pragma unroll
  for (int t = 0; t < 2; ++t) {
    v8f acc = t ? acc1 : acc0;
    const int n = n0 + t;
    #pragma unroll
    for (int r = 0; r < 8; ++r) {
      const int M = r + 8 * g;     // row (token / key) within the 16-row tile
      float f = acc[r];
      size_t addr;
      if (m == 0) {                // q: A-layout tile (b, qi, hchunk c)
        f *= qscale;
        const int c = n >> 1;
        const int K = ((n & 1) << 4) | Nl;          // h % 32
        addr = (((size_t)(b * 128 + ti) * 4 + c) << 9) + laneA(M, K) * 16 + jA(K);
      } else if (m == 1) {         // k: B-layout tile, K = h%32, N = key
        const int c = n >> 1;
        const int K = ((n & 1) << 4) | Nl;
        addr = KOFF + (((size_t)(b * 128 + ti) * 4 + c) << 9) + laneB(K, M) * 16 + jB(K);
      } else {                     // v: B-layout tile, K = key%32, N = h%16
        const int K = M + ((ti & 1) << 4);
        addr = VOFF + (((size_t)(b * 64 + (ti >> 1)) * 8 + n) << 9) + laneB(K, Nl) * 16 + jB(K);
      }
      qkv[addr] = (__bf16)f;
    }
  }
}

// ---------------------------------------------------------------------------
// Phase 2: flash attention. One wave per (batch, 16-row q tile).
// grid = 256 blocks * 4 waves = 1024 tiles.
// ---------------------------------------------------------------------------
__global__ void __launch_bounds__(128)
attn_kernel(const __bf16* __restrict__ qkv, float* __restrict__ y) {
  __shared__ __bf16 pbuf[4][TILE_ELEMS];       // per-wave P transpose buffer

  const int lane = threadIdx.x & 31;
  const int wave = threadIdx.x >> 5;
  const int wid  = blockIdx.x * 4 + wave;      // 0..1023
  const int b    = wid >> 7;
  const int qi   = wid & 127;
  const int g    = lane >> 4;
  const int Nl   = lane & 15;

  const __bf16* kbase = qkv + KOFF;
  const __bf16* vbase = qkv + VOFF;

  // Load the full 16x128 q tile (4 A-layout chunks), contiguous per lane.
  v16bf qa[4];
  #pragma unroll
  for (int c = 0; c < 4; ++c)
    qa[c] = *(const v16bf*)(qkv + (((size_t)(b * 128 + qi) * 4 + c) << 9) + lane * 16);

  float mrow[8], lrow[8];
  v8f o[8];
  #pragma unroll
  for (int r = 0; r < 8; ++r) { mrow[r] = -1e30f; lrow[r] = 0.f; }
  #pragma unroll
  for (int n = 0; n < 8; ++n) o[n] = (v8f){};

  const int jmax = qi >> 1;
  for (int j = 0; j <= jmax; ++j) {
    // Prefetch next chunk's k/v tiles into cache (global_prefetch_b8).
    if (j < jmax) {
      __builtin_prefetch(kbase + (((size_t)(b * 128 + 2 * (j + 1)) * 4) << 9) + lane * 16, 0, 0);
      __builtin_prefetch(vbase + (((size_t)(b * 64 + (j + 1)) * 8) << 9) + lane * 16, 0, 0);
    }

    // ---- S = q @ k^T for 32 keys (two 16-key sub-tiles) ----
    v8f s0 = {}, s1 = {};
    #pragma unroll
    for (int c = 0; c < 4; ++c) {
      v16bf k0 = *(const v16bf*)(kbase + (((size_t)(b * 128 + 2 * j + 0) * 4 + c) << 9) + lane * 16);
      v16bf k1 = *(const v16bf*)(kbase + (((size_t)(b * 128 + 2 * j + 1) * 4 + c) << 9) + lane * 16);
      s0 = wmma_bf16(qa[c], k0, s0);
      s1 = wmma_bf16(qa[c], k1, s1);
    }

    // ---- causal mask (diagonal chunk only) + per-lane row max ----
    const bool last = (j == jmax);
    float cmx[8];
    #pragma unroll
    for (int r = 0; r < 8; ++r) {
      const int row = (qi << 4) + r + 8 * g;   // global token row
      const int k0i = 32 * j + Nl;             // key of s0 element
      const int k1i = 32 * j + 16 + Nl;        // key of s1 element
      float e0 = s0[r], e1 = s1[r];
      if (last) {
        if (k0i > row) e0 = -1e30f;
        if (k1i > row) e1 = -1e30f;
      }
      s0[r] = e0; s1[r] = e1;
      cmx[r] = fmaxf(e0, e1);
    }
    // reduce row max across the 16 lanes holding each row
    #pragma unroll
    for (int d = 1; d < 16; d <<= 1) {
      #pragma unroll
      for (int r = 0; r < 8; ++r)
        cmx[r] = fmaxf(cmx[r], __shfl_xor(cmx[r], d, 32));
    }

    // ---- online softmax update ----
    float scal[8], rs[8];
    #pragma unroll
    for (int r = 0; r < 8; ++r) {
      const float newm = fmaxf(mrow[r], cmx[r]);
      scal[r] = __expf(mrow[r] - newm);
      mrow[r] = newm;
      const float p0 = __expf(s0[r] - newm);
      const float p1 = __expf(s1[r] - newm);
      s0[r] = p0; s1[r] = p1;
      rs[r] = p0 + p1;
    }
    #pragma unroll
    for (int d = 1; d < 16; d <<= 1) {
      #pragma unroll
      for (int r = 0; r < 8; ++r)
        rs[r] += __shfl_xor(rs[r], d, 32);
    }
    #pragma unroll
    for (int r = 0; r < 8; ++r) lrow[r] = lrow[r] * scal[r] + rs[r];
    #pragma unroll
    for (int n = 0; n < 8; ++n)
      #pragma unroll
      for (int r = 0; r < 8; ++r) o[n][r] *= scal[r];

    // ---- P: D-layout f32 -> A-layout bf16 through per-wave LDS ----
    #pragma unroll
    for (int r = 0; r < 8; ++r) {
      const int M = r + 8 * g;
      { const int K = Nl;      pbuf[wave][laneA(M, K) * 16 + jA(K)] = (__bf16)s0[r]; }
      { const int K = 16 | Nl; pbuf[wave][laneA(M, K) * 16 + jA(K)] = (__bf16)s1[r]; }
    }
    asm volatile("s_wait_dscnt 0x0" ::: "memory");   // all ds stores visible
    v16bf p = *(const v16bf*)&pbuf[wave][lane * 16];
    asm volatile("" ::: "memory");                   // keep load before next stores

    // ---- O += P @ V (K = 32 keys) over all 8 h-tiles ----
    #pragma unroll
    for (int n = 0; n < 8; ++n) {
      v16bf vt = *(const v16bf*)(vbase + (((size_t)(b * 64 + j) * 8 + n) << 9) + lane * 16);
      o[n] = wmma_bf16(p, vt, o[n]);
    }
  }

  // ---- epilogue: normalize and store y[b, t, h] (f32) ----
  float inv[8];
  #pragma unroll
  for (int r = 0; r < 8; ++r) inv[r] = 1.0f / lrow[r];
  #pragma unroll
  for (int n = 0; n < 8; ++n) {
    #pragma unroll
    for (int r = 0; r < 8; ++r) {
      const int t = (qi << 4) + r + 8 * g;
      y[((size_t)(b * Tq + t)) * Hq + n * 16 + Nl] = o[n][r] * inv[r];
    }
  }
}

// ---------------------------------------------------------------------------
extern "C" void kernel_launch(void* const* d_in, const int* in_sizes, int n_in,
                              void* d_out, int out_size, void* d_ws, size_t ws_size,
                              hipStream_t stream) {
  const float* x  = (const float*)d_in[0];
  const float* Wq = (const float*)d_in[1];
  const float* Wk = (const float*)d_in[2];
  const float* Wv = (const float*)d_in[3];
  float* y = (float*)d_out;
  __bf16* qkv = (__bf16*)d_ws;   // needs 12.75 MiB

  wprep_kernel<<<192, 128, 0, stream>>>(Wq, Wk, Wv, qkv);
  dim3 g1(1024, 3);
  proj_kernel<<<g1, 128, 0, stream>>>(x, qkv);
  attn_kernel<<<256, 128, 0, stream>>>(qkv, y);
}
